// AttentionNewSVD_36644660970255
// MI455X (gfx1250) — matrix operand order repaired and checked by reading, not compiled, not run on hardware
//
#include <hip/hip_runtime.h>

// ---------------------------------------------------------------------------
// AttentionNewSVD for MI455X (gfx1250, wave32, WMMA 16x16x32 bf16)
//   B=4, N=2048, C=768, H=12, r=32, scale = 1/8 (head_dim 64)
// ---------------------------------------------------------------------------

#define Bq   4
#define Nq   2048
#define Cq   768
#define Hq   12
#define Rq   32
#define NHq  (Hq * Rq)      // 384
#define Mq   (Bq * Nq)      // 8192
#define SCALEF 0.125f       // 64^-0.5

typedef __attribute__((ext_vector_type(16))) __bf16 v16bf;
typedef __attribute__((ext_vector_type(8)))  __bf16 v8bf;
typedef __attribute__((ext_vector_type(8)))  float  v8f;

union V16U { v16bf v; v8bf h[2]; };

// native bf16 convert (gfx1250 has hw bf16 cvt; RTNE)
__device__ __forceinline__ __bf16 f32_to_bf16(float f) { return (__bf16)f; }

// --------------------------- small prep kernels ----------------------------

__global__ __launch_bounds__(256) void convert_bf16_kernel(
    __bf16* __restrict__ dst, const float* __restrict__ src, int n) {
    int i = blockIdx.x * 256 + threadIdx.x;
    if (i < n) dst[i] = f32_to_bf16(src[i]);
}

// dst[n*K + k] = bf16(src[k*N + n])   (weight transpose+convert, src is [K,N])
__global__ __launch_bounds__(256) void transpose_convert_kernel(
    __bf16* __restrict__ dst, const float* __restrict__ src, int K, int Nn) {
    int i = blockIdx.x * 256 + threadIdx.x;
    if (i >= K * Nn) return;
    int n = i / K;
    int k = i - n * K;
    dst[i] = f32_to_bf16(src[(size_t)k * Nn + n]);
}

// ------------------------------- WMMA GEMM ---------------------------------
// C = A[M,K] (bf16, row major) @ BT[N,K]^T (bf16).  256 thr = 8 waves,
// wave tile 16Mx64N, block tile 64Mx128N.  M%64==0, N%128==0, K%32==0.
// MODE 0: f32 row-major [M,N] -> Cf
// MODE 1: bf16 row-major [M,N] -> Cb
// MODE 2: bf16 V-transposed [(b*H+h)*32+r][npos] -> Cb, packed b128 stores
template <int MODE>
__global__ __launch_bounds__(256) void gemm_bf16_kernel(
    const __bf16* __restrict__ A, const __bf16* __restrict__ BT,
    __bf16* __restrict__ Cb, float* __restrict__ Cf,
    int M, int Nn, int K, float scale) {
    const int lane = threadIdx.x & 31;
    const int wave = threadIdx.x >> 5;
    const int wm = wave & 3, wn = wave >> 2;
    const int m0 = blockIdx.x * 64 + wm * 16;
    const int n0 = blockIdx.y * 128 + wn * 64;
    const int l16 = lane & 15, hl = lane >> 4;

    v8f acc[4] = {};
    const __bf16* arow = A + (size_t)(m0 + l16) * K;

    for (int kk = 0; kk < K; kk += 32) {
        __builtin_prefetch(arow + kk + 128, 0, 1);   // global_prefetch next tiles
        V16U a;
        a.h[0] = *(const v8bf*)(arow + kk + 16 * hl);       // K = 16hl..16hl+7
        a.h[1] = *(const v8bf*)(arow + kk + 16 * hl + 8);   // K = 16hl+8..16hl+15
#pragma unroll
        for (int t = 0; t < 4; ++t) {
            const __bf16* brow =
                BT + (size_t)(n0 + t * 16 + l16) * K + kk + 16 * hl;
            v16bf b = *(const v16bf*)brow;                  // K = kk+16hl .. +16
            acc[t] = __builtin_amdgcn_wmma_f32_16x16x32_bf16(
                false, a.v, false, b, (short)0, acc[t], false, false);
        }
    }

    if (MODE == 2) {
        // direct VT emission: for fixed col, rows g=0..7 are consecutive
        // sequence positions -> one packed 16B store per tile column.
        const int bidx = m0 >> 11;                 // batch (tiles never cross)
        const int npos = (m0 & (Nq - 1)) + 8 * hl; // 16B aligned
#pragma unroll
        for (int t = 0; t < 4; ++t) {
            int col = n0 + t * 16 + l16;           // h*32 + r
            int hh = col >> 5, rr = col & 31;
            v8bf pk;
#pragma unroll
            for (int g = 0; g < 8; ++g) pk[g] = f32_to_bf16(acc[t][g] * scale);
            *(v8bf*)(Cb + ((size_t)(bidx * Hq + hh) * 32 + rr) * Nq + npos) = pk;
        }
    } else {
#pragma unroll
        for (int t = 0; t < 4; ++t) {
#pragma unroll
            for (int g = 0; g < 8; ++g) {
                int row = m0 + g + 8 * hl;      // C layout: lanes>=16 -> M=8+g
                int col = n0 + t * 16 + l16;
                float v = acc[t][g] * scale;
                if (MODE == 1) Cb[(size_t)row * Nn + col] = f32_to_bf16(v);
                else           Cf[(size_t)row * Nn + col] = v;
            }
        }
    }
}

// --------------------------- fused flash attention -------------------------
// grid (N/128, H, B); 8 waves, each wave owns a 16-row Q tile and streams
// keys 64 at a time: 4 S-WMMAs -> one batched online-softmax round
// (max via shfl butterfly, SUM via WMMA against an all-ones B) -> P through
// wave-private LDS -> 4 P@V WMMAs.  Never materializes attn in HBM.
__global__ __launch_bounds__(256) void attn_kernel(
    const __bf16* __restrict__ Q, const __bf16* __restrict__ Kt,
    const __bf16* __restrict__ VT, __bf16* __restrict__ O) {
    __shared__ __bf16 lds_p[8][16][64];   // per-wave P slab, 16 KB total

    const int lane = threadIdx.x & 31;
    const int wave = threadIdx.x >> 5;
    const int l16 = lane & 15, hl = lane >> 4;
    const int qb = blockIdx.x;            // q block (128 rows)
    const int h  = blockIdx.y;
    const int b  = blockIdx.z;
    const int q0 = qb * 128 + wave * 16;

    // Q A-fragment (K = r = 32 exactly), held for the whole kernel
    V16U aq;
    {
        const __bf16* qrow = Q + (size_t)(b * Nq + q0 + l16) * NHq + h * Rq;
        aq.h[0] = *(const v8bf*)(qrow + 16 * hl);
        aq.h[1] = *(const v8bf*)(qrow + 16 * hl + 8);
    }

    // all-ones B fragment for rowsum-by-WMMA
    v16bf ones;
#pragma unroll
    for (int e = 0; e < 16; ++e)
        ones[e] = __builtin_bit_cast(__bf16, (unsigned short)0x3F80);

    v8f acc0 = {}, acc1 = {};             // O tile 16 x 32 (two 16-col halves)
    v8f accS = {};                        // running softmax denominators
    float mrow[8];
#pragma unroll
    for (int g = 0; g < 8; ++g) mrow[g] = -__builtin_inff();

    const __bf16* kbase = Kt + (size_t)b * Nq * NHq + h * Rq + 16 * hl;
    const __bf16* vbase = VT + ((size_t)(b * Hq + h) * 32 + l16) * Nq + 16 * hl;

    for (int j = 0; j < Nq; j += 64) {
        __builtin_prefetch(kbase + (size_t)(j + 64 + l16) * NHq, 0, 1);
        __builtin_prefetch(vbase + j + 64, 0, 1);

        // ---- S = Q @ K^T for keys [j, j+64) : four 16x16 tiles ----
        v8f s[4];
#pragma unroll
        for (int t = 0; t < 4; ++t) {
            v16bf bk = *(const v16bf*)(kbase + (size_t)(j + t * 16 + l16) * NHq);
            v8f z = {};
            s[t] = __builtin_amdgcn_wmma_f32_16x16x32_bf16(
                false, aq.v, false, bk, (short)0, z, false, false);
        }

        // ---- batched online softmax: one max-butterfly per 64 keys ----
#pragma unroll
        for (int g = 0; g < 8; ++g) {
            float t = fmaxf(fmaxf(s[0][g], s[1][g]), fmaxf(s[2][g], s[3][g]));
#pragma unroll
            for (int off = 1; off < 16; off <<= 1)
                t = fmaxf(t, __shfl_xor(t, off, 16));
            float mnew  = fmaxf(mrow[g], t);
            float alpha = __expf(mrow[g] - mnew);
            mrow[g] = mnew;
            acc0[g] *= alpha;
            acc1[g] *= alpha;
            accS[g] *= alpha;
            int row = g + 8 * hl;                 // C layout row for this lane
#pragma unroll
            for (int t4 = 0; t4 < 4; ++t4)
                lds_p[wave][row][t4 * 16 + l16] =
                    f32_to_bf16(__expf(s[t4][g] - mnew));
        }

        // ---- P (C-layout f32) -> two A-fragments via wave-private LDS ----
        V16U ap0, ap1;                            // keys j..j+31, j+32..j+63
        ap0.h[0] = *(const v8bf*)(&lds_p[wave][l16][16 * hl]);
        ap0.h[1] = *(const v8bf*)(&lds_p[wave][l16][16 * hl + 8]);
        ap1.h[0] = *(const v8bf*)(&lds_p[wave][l16][32 + 16 * hl]);
        ap1.h[1] = *(const v8bf*)(&lds_p[wave][l16][40 + 16 * hl]);

        // ---- denominators: rowsum(P) = P @ ones, no cross-lane shuffles ----
        accS = __builtin_amdgcn_wmma_f32_16x16x32_bf16(
            false, ap0.v, false, ones, (short)0, accS, false, false);
        accS = __builtin_amdgcn_wmma_f32_16x16x32_bf16(
            false, ap1.v, false, ones, (short)0, accS, false, false);

        // ---- O += P @ V (B fragments contiguous from pre-transposed VT) ----
        v16bf bv00 = *(const v16bf*)(vbase + j);                        // K j..  , r 0..15
        v16bf bv01 = *(const v16bf*)(vbase + (size_t)16 * Nq + j);      // K j..  , r 16..31
        v16bf bv10 = *(const v16bf*)(vbase + j + 32);                   // K j+32., r 0..15
        v16bf bv11 = *(const v16bf*)(vbase + (size_t)16 * Nq + j + 32); // K j+32., r 16..31
        acc0 = __builtin_amdgcn_wmma_f32_16x16x32_bf16(
            false, ap0.v, false, bv00, (short)0, acc0, false, false);
        acc1 = __builtin_amdgcn_wmma_f32_16x16x32_bf16(
            false, ap0.v, false, bv01, (short)0, acc1, false, false);
        acc0 = __builtin_amdgcn_wmma_f32_16x16x32_bf16(
            false, ap1.v, false, bv10, (short)0, acc0, false, false);
        acc1 = __builtin_amdgcn_wmma_f32_16x16x32_bf16(
            false, ap1.v, false, bv11, (short)0, acc1, false, false);
    }

    // normalize and store [B*N, H*32] bf16 (accS[g] uniform across each half)
#pragma unroll
    for (int g = 0; g < 8; ++g) {
        int row = q0 + g + 8 * hl;
        float inv = 1.0f / accS[g];
        __bf16* op = O + (size_t)(b * Nq + row) * NHq + h * Rq;
        op[l16]      = f32_to_bf16(acc0[g] * inv);
        op[16 + l16] = f32_to_bf16(acc1[g] * inv);
    }
}

// ------------------------------- launcher ----------------------------------

extern "C" void kernel_launch(void* const* d_in, const int* in_sizes, int n_in,
                              void* d_out, int out_size, void* d_ws, size_t ws_size,
                              hipStream_t stream) {
    const float* x  = (const float*)d_in[0];
    const float* Wq = (const float*)d_in[1];
    const float* Wk = (const float*)d_in[2];
    const float* Wv = (const float*)d_in[3];
    const float* Wp = (const float*)d_in[4];
    float* out = (float*)d_out;

    char* ws = (char*)d_ws;
    size_t off = 0;
    auto alloc = [&](size_t bytes) -> char* {
        char* p = ws + off;
        off = (off + bytes + 255) & ~(size_t)255;
        return p;
    };
    __bf16* xb  = (__bf16*)alloc((size_t)Mq * Cq * 2);     // x bf16
    __bf16* wqT = (__bf16*)alloc((size_t)NHq * Cq * 2);    // [384,768]
    __bf16* wkT = (__bf16*)alloc((size_t)NHq * Cq * 2);
    __bf16* wvT = (__bf16*)alloc((size_t)NHq * Cq * 2);
    __bf16* wpT = (__bf16*)alloc((size_t)Cq * NHq * 2);    // [768,384]
    __bf16* qb  = (__bf16*)alloc((size_t)Mq * NHq * 2);
    __bf16* kb  = (__bf16*)alloc((size_t)Mq * NHq * 2);
    __bf16* vtb = (__bf16*)alloc((size_t)Mq * NHq * 2);    // V in [b,h,r][n]
    __bf16* aob = (__bf16*)alloc((size_t)Mq * NHq * 2);

    // 1) precision conversions / weight transposes
    convert_bf16_kernel<<<(Mq * Cq + 255) / 256, 256, 0, stream>>>(xb, x, Mq * Cq);
    transpose_convert_kernel<<<(Cq * NHq + 255) / 256, 256, 0, stream>>>(wqT, Wq, Cq, NHq);
    transpose_convert_kernel<<<(Cq * NHq + 255) / 256, 256, 0, stream>>>(wkT, Wk, Cq, NHq);
    transpose_convert_kernel<<<(Cq * NHq + 255) / 256, 256, 0, stream>>>(wvT, Wv, Cq, NHq);
    transpose_convert_kernel<<<(NHq * Cq + 255) / 256, 256, 0, stream>>>(wpT, Wp, NHq, Cq);

    // 2) q/k/v low-rank projections (WMMA); softmax scale folded into q;
    //    v written directly in per-head transposed layout (packed stores)
    dim3 gp(Mq / 64, NHq / 128);
    gemm_bf16_kernel<1><<<gp, 256, 0, stream>>>(xb, wqT, qb, nullptr, Mq, NHq, Cq, SCALEF);
    gemm_bf16_kernel<1><<<gp, 256, 0, stream>>>(xb, wkT, kb, nullptr, Mq, NHq, Cq, 1.0f);
    gemm_bf16_kernel<2><<<gp, 256, 0, stream>>>(xb, wvT, vtb, nullptr, Mq, NHq, Cq, 1.0f);

    // 3) fused flash attention (never materializes the 805 MB attn matrix)
    dim3 ga(Nq / 128, Hq, Bq);
    attn_kernel<<<ga, 256, 0, stream>>>(qb, kb, vtb, aob);

    // 4) output projection -> f32 d_out
    dim3 go(Mq / 64, Cq / 128);
    gemm_bf16_kernel<0><<<go, 256, 0, stream>>>(aob, wpT, nullptr, out, Mq, Cq, NHq, 1.0f);
}
